// Attention_17703855194428
// MI455X (gfx1250) — matrix operand order repaired and checked by reading, not compiled, not run on hardware
//
#include <hip/hip_runtime.h>

typedef __attribute__((ext_vector_type(16))) _Float16 v16h;
typedef __attribute__((ext_vector_type(8)))  float    v8f;

#define DIMC   256
#define HEADS  8
#define DH     32
#define AGENT  6
#define WINSZ  7
#define NTOK   294            // 6*7*7
#define NPAD   320            // padded token count (10 blocks of 32)
#define NWIN   128            // B*X*Y = 2*8*8

union AFrag { v16h v; uint4 q[2]; };
union HVec8 { uint4 q; _Float16 h[8]; };

// ---------------------------------------------------------------- setup ---

// x (B,L,X,Y,7,7,C) f32 -> xf16 [win][NPAD][C] f16, permuted, zero-padded rows
__global__ __launch_bounds__(256) void cvt_x_kernel(const float* __restrict__ x,
                                                    _Float16* __restrict__ xf) {
  long i = (long)blockIdx.x * 256 + threadIdx.x;      // over 128*320*256
  if (i >= (long)NWIN * NPAD * DIMC) return;
  int c = (int)(i & 255);
  long t = i >> 8;
  int n = (int)(t % NPAD);
  int win = (int)(t / NPAD);
  float val = 0.f;
  if (n < NTOK) {
    int b_ = win >> 6, xx = (win >> 3) & 7, yy = win & 7;
    int l = n / 49, r = n % 49, w1 = r / 7, w2 = r % 7;
    long src = (((((long)(b_ * AGENT + l) * 8 + xx) * 8 + yy) * WINSZ + w1) * WINSZ + w2) * DIMC + c;
    val = x[src];
  }
  xf[i] = (_Float16)val;
}

__global__ __launch_bounds__(256) void cvt_w_kernel(const float* __restrict__ w,
                                                    _Float16* __restrict__ wf, int n) {
  int i = blockIdx.x * 256 + threadIdx.x;
  if (i < n) wf[i] = (_Float16)w[i];
}

// bias_table (1859,8) -> biasPad [h][NPAD][NPAD] f32, zero outside valid tokens
__global__ __launch_bounds__(256) void make_bias_kernel(const float* __restrict__ bt,
                                                        float* __restrict__ bp) {
  long i = (long)blockIdx.x * 256 + threadIdx.x;      // over 8*320*320
  if (i >= (long)HEADS * NPAD * NPAD) return;
  int nk = (int)(i % NPAD);
  long t = i / NPAD;
  int nq = (int)(t % NPAD);
  int h  = (int)(t / NPAD);
  float v = 0.f;
  if (nq < NTOK && nk < NTOK) {
    int li = nq / 49, ri = nq % 49, w1i = ri / 7, w2i = ri % 7;
    int lj = nk / 49, rj = nk % 49, w1j = rj / 7, w2j = rj % 7;
    int idx = (li - lj + AGENT - 1) * 169 + (w1i - w1j + 6) * 13 + (w2i - w2j + 6);
    v = bt[idx * HEADS + h];
  }
  bp[i] = v;
}

// mask (B,X,Y,7,7,1,AGENT) i32 -> maskPad [win][NPAD] f32 in {0, -1e9}; pad = -1e9
__global__ __launch_bounds__(256) void make_mask_kernel(const int* __restrict__ m,
                                                        float* __restrict__ mp) {
  int i = blockIdx.x * 256 + threadIdx.x;             // over 128*320
  if (i >= NWIN * NPAD) return;
  int n = i % NPAD, win = i / NPAD;
  float v = -1e9f;
  if (n < NTOK) {
    int b_ = win >> 6, xx = (win >> 3) & 7, yy = win & 7;
    int l = n / 49, r = n % 49, w1 = r / 7, w2 = r % 7;
    long src = ((long)(((b_ * 8 + xx) * 8 + yy) * WINSZ + w1) * WINSZ + w2) * AGENT + l;
    v = (m[src] == 0) ? -1e9f : 0.f;
  }
  mp[i] = v;
}

// ------------------------------------------------------------- QKV GEMM ---
// qkv[r][e] = sum_c xf16[r][c] * wqkv[e][c];  M = 128*320 rows, N = 768, K = 256
// One 16x16 output tile per wave; 8 waves / block.
__global__ __launch_bounds__(256) void qkv_gemm_kernel(const _Float16* __restrict__ xf,
                                                       const _Float16* __restrict__ wq,
                                                       _Float16* __restrict__ qb,
                                                       _Float16* __restrict__ kb,
                                                       _Float16* __restrict__ vT) {
  int wave = threadIdx.x >> 5, ln = threadIdx.x & 31;
  int lane16 = ln & 15, half = ln >> 4;
  long tid = (long)blockIdx.x * 8 + wave;             // global tile id
  int etile = (int)(tid % 48);                        // 48 e-tiles of 16 (3*256/16)
  int mtile = (int)(tid / 48);                        // 0..2559
  int win = mtile / 20, rt = mtile % 20;

  const _Float16* arow = xf + ((long)win * NPAD + rt * 16 + lane16) * DIMC;
  const _Float16* brow = wq + (long)(etile * 16 + lane16) * DIMC + half * 16;

  v8f acc = {};
  for (int c0 = 0; c0 < DIMC; c0 += 32) {
    AFrag a;
    a.q[0] = *(const uint4*)(arow + c0 + half * 16);       // K = half*16 + 0..7
    a.q[1] = *(const uint4*)(arow + c0 + half * 16 + 8);   // K = half*16 + 8..15
    v16h b = *(const v16h*)(brow + c0);                    // K = half*16 + 0..15
    acc = __builtin_amdgcn_wmma_f32_16x16x32_f16(false, a.v, false, b,
                                                 (short)0, acc, false, false);
  }

  int mat   = etile / 16;              // 0=q 1=k 2=v
  int head  = (etile >> 1) & 7;
  int dbase = (etile & 1) * 16 + lane16;     // N = lane16
  int nbase = rt * 16 + half * 8;            // M = v + 8*half

  if (mat == 2) {                            // V stored transposed [win][h][32][NPAD]
    HVec8 hv;
    for (int v = 0; v < 8; ++v) hv.h[v] = (_Float16)acc[v];
    *(uint4*)(vT + ((long)(win * HEADS + head) * DH + dbase) * NPAD + nbase) = hv.q;
  } else {
    float scl = (mat == 0) ? 0.17677669529663689f : 1.0f;  // 1/sqrt(32) on Q
    _Float16* dst = (mat == 0 ? qb : kb) +
                    ((long)(win * HEADS + head) * NPAD + nbase) * DH + dbase;
    for (int v = 0; v < 8; ++v) dst[v * DH] = (_Float16)(acc[v] * scl);
  }
}

// ------------------------------------------------------- flash attention ---
// One block per (window, head). K and V^T staged in LDS; each wave owns a
// 16-row Q tile with online softmax over 10 key-blocks of 32.
__global__ __launch_bounds__(256) void attn_kernel(const _Float16* __restrict__ qb,
                                                   const _Float16* __restrict__ kb,
                                                   const _Float16* __restrict__ vT,
                                                   const float* __restrict__ biasPad,
                                                   const float* __restrict__ maskPad,
                                                   _Float16* __restrict__ outPre) {
  __shared__ __align__(32) _Float16 lds_k[NPAD * DH];      // 20480 B
  __shared__ __align__(32) _Float16 lds_v[DH * NPAD];      // 20480 B
  __shared__ __align__(32) _Float16 pstage[8][16 * 32];    //  8192 B

  int bid = blockIdx.x;                      // 0..1023
  int win = bid >> 3, head = bid & 7;
  int tid = threadIdx.x;

  { // cooperative stage of K and V^T for this (win, head)
    const uint4* sk = (const uint4*)(kb + (long)(win * HEADS + head) * NPAD * DH);
    const uint4* sv = (const uint4*)(vT + (long)(win * HEADS + head) * DH * NPAD);
    uint4* dk = (uint4*)lds_k; uint4* dv = (uint4*)lds_v;
    for (int i = tid; i < NPAD * DH / 8; i += 256) { dk[i] = sk[i]; dv[i] = sv[i]; }
  }
  __syncthreads();

  int wave = tid >> 5, ln = tid & 31;
  int lane16 = ln & 15, half = ln >> 4;
  const float* biasH = biasPad + (long)head * NPAD * NPAD;
  const float* maskW = maskPad + win * NPAD;

  for (int qt = wave; qt < 19; qt += 8) {    // 19 tiles cover rows 0..303
    int q0 = qt * 16;
    AFrag qa;
    const _Float16* qrow = qb + ((long)(win * HEADS + head) * NPAD + q0 + lane16) * DH;
    qa.q[0] = *(const uint4*)(qrow + half * 16);
    qa.q[1] = *(const uint4*)(qrow + half * 16 + 8);

    v8f o0 = {}, o1 = {};
    float mrow[8], lrow[8];
    for (int v = 0; v < 8; ++v) { mrow[v] = -1e30f; lrow[v] = 0.f; }

    for (int k0 = 0; k0 < NPAD; k0 += 32) {
      v16h bk0 = *(const v16h*)(lds_k + (k0 + lane16) * DH + half * 16);
      v16h bk1 = *(const v16h*)(lds_k + (k0 + 16 + lane16) * DH + half * 16);
      v8f z = {};
      v8f s0 = __builtin_amdgcn_wmma_f32_16x16x32_f16(false, qa.v, false, bk0,
                                                      (short)0, z, false, false);
      v8f s1 = __builtin_amdgcn_wmma_f32_16x16x32_f16(false, qa.v, false, bk1,
                                                      (short)0, z, false, false);
      float mk0 = maskW[k0 + lane16];
      float mk1 = maskW[k0 + 16 + lane16];
      const float* bp = biasH + (long)(q0 + 8 * half) * NPAD + k0;
      float sa[8], sb[8];
      for (int v = 0; v < 8; ++v) {
        const float* br = bp + v * NPAD;
        sa[v] = s0[v] + br[lane16] + mk0;
        sb[v] = s1[v] + br[16 + lane16] + mk1;
      }
      // online softmax: per-row (16-lane half) max / sum
      float scale[8];
      for (int v = 0; v < 8; ++v) {
        float r = fmaxf(sa[v], sb[v]);
        r = fmaxf(r, __shfl_xor(r, 8, 16));
        r = fmaxf(r, __shfl_xor(r, 4, 16));
        r = fmaxf(r, __shfl_xor(r, 2, 16));
        r = fmaxf(r, __shfl_xor(r, 1, 16));
        float mnew = fmaxf(mrow[v], r);
        scale[v] = __expf(mrow[v] - mnew);
        mrow[v] = mnew;
      }
      for (int v = 0; v < 8; ++v) {
        sa[v] = __expf(sa[v] - mrow[v]);
        sb[v] = __expf(sb[v] - mrow[v]);
        float r = sa[v] + sb[v];
        r += __shfl_xor(r, 8, 16);
        r += __shfl_xor(r, 4, 16);
        r += __shfl_xor(r, 2, 16);
        r += __shfl_xor(r, 1, 16);
        lrow[v] = lrow[v] * scale[v] + r;
        o0[v] *= scale[v];
        o1[v] *= scale[v];
      }
      // C-layout -> A-layout transpose of P through per-wave LDS stage
      _Float16* ps = pstage[wave];
      for (int v = 0; v < 8; ++v) {
        ps[(v + 8 * half) * 32 + lane16]      = (_Float16)sa[v];
        ps[(v + 8 * half) * 32 + 16 + lane16] = (_Float16)sb[v];
      }
      asm volatile("s_wait_dscnt 0x0" ::: "memory");
      AFrag pf;
      pf.q[0] = *(const uint4*)(ps + lane16 * 32 + half * 16);
      pf.q[1] = *(const uint4*)(ps + lane16 * 32 + half * 16 + 8);
      v16h bv0 = *(const v16h*)(lds_v + (long)lane16 * NPAD + k0 + half * 16);
      v16h bv1 = *(const v16h*)(lds_v + (long)(16 + lane16) * NPAD + k0 + half * 16);
      o0 = __builtin_amdgcn_wmma_f32_16x16x32_f16(false, pf.v, false, bv0,
                                                  (short)0, o0, false, false);
      o1 = __builtin_amdgcn_wmma_f32_16x16x32_f16(false, pf.v, false, bv1,
                                                  (short)0, o1, false, false);
    }
    // epilogue: divide by l, store to outPre [win][NPAD][C] (pad rows -> 0)
    _Float16* orow = outPre + ((long)win * NPAD + q0 + 8 * half) * DIMC + head * DH;
    for (int v = 0; v < 8; ++v) {
      int n = q0 + v + 8 * half;
      float inv = 1.0f / lrow[v];
      float a = (n < NTOK) ? o0[v] * inv : 0.f;
      float b = (n < NTOK) ? o1[v] * inv : 0.f;
      orow[v * DIMC + lane16]      = (_Float16)a;
      orow[v * DIMC + 16 + lane16] = (_Float16)b;
    }
  }
}

// ------------------------------------------------------- output projection ---
// out[r][e] = sum_c outPre[r][c] * w_out[e][c], scattered to (B,L,X,Y,7,7,C) f32
__global__ __launch_bounds__(256) void out_gemm_kernel(const _Float16* __restrict__ op,
                                                       const _Float16* __restrict__ wo,
                                                       float* __restrict__ out) {
  int wave = threadIdx.x >> 5, ln = threadIdx.x & 31;
  int lane16 = ln & 15, half = ln >> 4;
  long tid = (long)blockIdx.x * 8 + wave;
  int etile = (int)(tid % 16);
  int mtile = (int)(tid / 16);               // 0..2431
  int win = mtile / 19, rt = mtile % 19;

  const _Float16* arow = op + ((long)win * NPAD + rt * 16 + lane16) * DIMC;
  const _Float16* brow = wo + (long)(etile * 16 + lane16) * DIMC + half * 16;

  v8f acc = {};
  for (int c0 = 0; c0 < DIMC; c0 += 32) {
    AFrag a;
    a.q[0] = *(const uint4*)(arow + c0 + half * 16);
    a.q[1] = *(const uint4*)(arow + c0 + half * 16 + 8);
    v16h b = *(const v16h*)(brow + c0);
    acc = __builtin_amdgcn_wmma_f32_16x16x32_f16(false, a.v, false, b,
                                                 (short)0, acc, false, false);
  }

  int b_ = win >> 6, xx = (win >> 3) & 7, yy = win & 7;
  int e = etile * 16 + lane16;
  for (int v = 0; v < 8; ++v) {
    int nn = rt * 16 + v + 8 * half;
    if (nn < NTOK) {
      int l = nn / 49, r49 = nn % 49, w1 = r49 / 7, w2 = r49 % 7;
      long off = (((((long)(b_ * AGENT + l) * 8 + xx) * 8 + yy) * WINSZ + w1) * WINSZ + w2) * DIMC + e;
      out[off] = acc[v];
    }
  }
}

// ----------------------------------------------------------------- launch ---
extern "C" void kernel_launch(void* const* d_in, const int* in_sizes, int n_in,
                              void* d_out, int out_size, void* d_ws, size_t ws_size,
                              hipStream_t stream) {
  const float* x    = (const float*)d_in[0];
  const int*   mask = (const int*)d_in[1];
  const float* wqkv = (const float*)d_in[2];
  const float* wout = (const float*)d_in[3];
  const float* btab = (const float*)d_in[4];

  char* ws = (char*)d_ws;
  constexpr long SZ_XF   = (long)NWIN * NPAD * DIMC * 2;       // 20,971,520
  constexpr long SZ_QKV  = (long)NWIN * HEADS * NPAD * DH * 2; // 20,971,520
  constexpr long SZ_OP   = (long)NWIN * NPAD * DIMC * 2;       // 20,971,520
  constexpr long SZ_WQ   = 768 * 256 * 2;
  constexpr long SZ_WO   = 256 * 256 * 2;
  constexpr long SZ_BIAS = (long)HEADS * NPAD * NPAD * 4;
  _Float16* xf   = (_Float16*)(ws);
  _Float16* qb   = (_Float16*)(ws + SZ_XF);
  _Float16* kbuf = (_Float16*)(ws + SZ_XF + SZ_QKV);
  _Float16* vT   = (_Float16*)(ws + SZ_XF + 2 * SZ_QKV);
  _Float16* op   = (_Float16*)(ws + SZ_XF + 3 * SZ_QKV);
  _Float16* wq16 = (_Float16*)(ws + SZ_XF + 3 * SZ_QKV + SZ_OP);
  _Float16* wo16 = (_Float16*)(ws + SZ_XF + 3 * SZ_QKV + SZ_OP + SZ_WQ);
  float* biasPad = (float*)   (ws + SZ_XF + 3 * SZ_QKV + SZ_OP + SZ_WQ + SZ_WO);
  float* maskPad = (float*)   (ws + SZ_XF + 3 * SZ_QKV + SZ_OP + SZ_WQ + SZ_WO + SZ_BIAS);

  cvt_x_kernel<<<(NWIN * NPAD * DIMC + 255) / 256, 256, 0, stream>>>(x, xf);
  cvt_w_kernel<<<(768 * 256 + 255) / 256, 256, 0, stream>>>(wqkv, wq16, 768 * 256);
  cvt_w_kernel<<<(256 * 256 + 255) / 256, 256, 0, stream>>>(wout, wo16, 256 * 256);
  make_bias_kernel<<<(HEADS * NPAD * NPAD + 255) / 256, 256, 0, stream>>>(btab, biasPad);
  make_mask_kernel<<<(NWIN * NPAD + 255) / 256, 256, 0, stream>>>(mask, maskPad);

  // 128 windows * 20 row-tiles * 48 e-tiles / 8 waves = 15360 blocks
  qkv_gemm_kernel<<<15360, 256, 0, stream>>>(xf, wq16, qb, kbuf, vT);
  // one block per (window, head)
  attn_kernel<<<NWIN * HEADS, 256, 0, stream>>>(qb, kbuf, vT, biasPad, maskPad, op);
  // 128 windows * 19 row-tiles * 16 e-tiles / 8 waves = 4864 blocks
  out_gemm_kernel<<<4864, 256, 0, stream>>>(op, wo16, (float*)d_out);
}
